// GlobalContextInjection_81432579932852
// MI455X (gfx1250) — matrix operand order repaired and checked by reading, not compiled, not run on hardware
//
#include <hip/hip_runtime.h>
#include <hip/hip_bf16.h>
#include <cstdint>

#define H   256
#define HH  128   // H/2
#define G   512

typedef __attribute__((ext_vector_type(16))) __bf16 v16bf;
typedef __attribute__((ext_vector_type(8)))  __bf16 v8bf;
typedef __attribute__((ext_vector_type(8)))  float  v8f;

// -------- order-preserving float<->uint encoding for atomic max --------
__device__ __forceinline__ unsigned enc_f(float f) {
  unsigned b = __float_as_uint(f);
  return (b & 0x80000000u) ? ~b : (b | 0x80000000u);
}
__device__ __forceinline__ float dec_f(unsigned e) {
  unsigned b = (e & 0x80000000u) ? (e ^ 0x80000000u) : ~e;
  return __uint_as_float(b);
}

// -------- 0: zero per-graph stats (smax enc / denom / emb) --------
__global__ __launch_bounds__(256) void zero_kernel(float* p, int n) {
  int i = blockIdx.x * 256 + threadIdx.x;
  if (i < n) p[i] = 0.0f;
}

// -------- 1: gate s = tanh(x@W1+b1)@W2+b2  (bf16 WMMA, f32 accum) --------
__global__ __launch_bounds__(256)
void gate_kernel(const float* __restrict__ x, const float* __restrict__ W1,
                 const float* __restrict__ b1, const float* __restrict__ W2,
                 const float* __restrict__ b2, float* __restrict__ s_out,
                 int N, int ntiles) {
  // W1 (256x128) as bf16 in fragment-swizzled layout:
// element (k,n) -> frag f = (k>>5)*8 + (n>>4); index ((f*32 + ((k>>3)&1)*16 + (n&15))*16 + (k&7) + (((k>>4)&1)<<3))
  __shared__ __bf16 w1lds[64 * 32 * 16];  // 64 KB

  {
    int k = threadIdx.x;                       // source row of W1
    const float4* src = (const float4*)(W1 + k * HH);
    int khi = k >> 5, klo = k & 31;
    #pragma unroll
    for (int n16 = 0; n16 < 8; ++n16) {
      __bf16* dst = &w1lds[((khi * 8 + n16) * 32 + ((klo >> 3) & 1) * 16) * 16 + (klo & 7) + ((klo >> 4) << 3)];
      #pragma unroll
      for (int q = 0; q < 4; ++q) {
        float4 f = src[n16 * 4 + q];
        dst[(q * 4 + 0) * 16] = (__bf16)f.x;
        dst[(q * 4 + 1) * 16] = (__bf16)f.y;
        dst[(q * 4 + 2) * 16] = (__bf16)f.z;
        dst[(q * 4 + 3) * 16] = (__bf16)f.w;
      }
    }
  }
  __syncthreads();

  int wave = threadIdx.x >> 5;
  int lane = threadIdx.x & 31;
  int tile = blockIdx.x * 8 + wave;
  if (tile >= ntiles) return;

  int col  = lane & 15;        // N-column within 16-wide tile (also A row id)
  int half = lane >> 4;        // lane group: K offset +8 for A frag
  int rowIdx = tile * 16 + col;
  int rowClamped = rowIdx < N ? rowIdx : (N - 1);

  float b2v = b2[0];

  v8f acc[8];
  #pragma unroll
  for (int nt = 0; nt < 8; ++nt) {
    float bv = b1[nt * 16 + col];   // bias depends only on output column
    #pragma unroll
    for (int j = 0; j < 8; ++j) acc[nt][j] = bv;
  }

  const float* xr = x + (size_t)rowClamped * H + half * 8;
  #pragma unroll
  for (int ks = 0; ks < 8; ++ks) {
    const float* xk = xr + ks * 32;
    // A fragment: halves 0..7 = K base..base+7, halves 8..15 = K base+16..base+23
    float4 f0 = *(const float4*)(xk);
    float4 f1 = *(const float4*)(xk + 4);
    float4 f2 = *(const float4*)(xk + 16);
    float4 f3 = *(const float4*)(xk + 20);
    v16bf a;
    a[0]=(__bf16)f0.x;  a[1]=(__bf16)f0.y;  a[2]=(__bf16)f0.z;  a[3]=(__bf16)f0.w;
    a[4]=(__bf16)f1.x;  a[5]=(__bf16)f1.y;  a[6]=(__bf16)f1.z;  a[7]=(__bf16)f1.w;
    a[8]=(__bf16)f2.x;  a[9]=(__bf16)f2.y;  a[10]=(__bf16)f2.z; a[11]=(__bf16)f2.w;
    a[12]=(__bf16)f3.x; a[13]=(__bf16)f3.y; a[14]=(__bf16)f3.z; a[15]=(__bf16)f3.w;

    #pragma unroll
    for (int nt = 0; nt < 8; ++nt) {
      const __bf16* bp_ = &w1lds[((ks * 8 + nt) * 32 + lane) * 16];
      union { v16bf v; v8bf h[2]; } bu;
      bu.h[0] = *(const v8bf*)(bp_);
      bu.h[1] = *(const v8bf*)(bp_ + 8);
      acc[nt] = __builtin_amdgcn_wmma_f32_16x16x32_bf16(
          false, a, false, bu.v, (short)0, acc[nt], false, false);
    }
  }

  // tanh, multiply by W2 column weight, reduce across the 16 columns
  float part[8];
  #pragma unroll
  for (int j = 0; j < 8; ++j) part[j] = 0.0f;
  #pragma unroll
  for (int nt = 0; nt < 8; ++nt) {
    float w2v = W2[nt * 16 + col];
    #pragma unroll
    for (int j = 0; j < 8; ++j)
      part[j] += tanhf(acc[nt][j]) * w2v;
  }
  #pragma unroll
  for (int m = 1; m < 16; m <<= 1) {
    #pragma unroll
    for (int j = 0; j < 8; ++j)
      part[j] += __shfl_xor(part[j], m, 32);
  }
  if (col == 0) {
    #pragma unroll
    for (int j = 0; j < 8; ++j) {
      int orow = tile * 16 + half * 8 + j;   // C layout: VGPR j -> row j + 8*half
      if (orow < N) s_out[orow] = part[j] + b2v;
    }
  }
}

// -------- 2: per-graph max --------
__global__ __launch_bounds__(256)
void segmax_kernel(const float* __restrict__ s, const long long* __restrict__ batch,
                   unsigned* __restrict__ smax, int N) {
  int i = blockIdx.x * 256 + threadIdx.x;
  if (i < N) atomicMax(&smax[(int)batch[i]], enc_f(s[i]));
}

// -------- 3: e = exp(s - smax), denom += e --------
__global__ __launch_bounds__(256)
void expsum_kernel(const float* __restrict__ s, const long long* __restrict__ batch,
                   const unsigned* __restrict__ smax, float* __restrict__ e,
                   float* __restrict__ denom, int N) {
  int i = blockIdx.x * 256 + threadIdx.x;
  if (i < N) {
    int g = (int)batch[i];
    float ev = expf(s[i] - dec_f(smax[g]));
    e[i] = ev;
    atomicAdd(&denom[g], ev);
  }
}

// -------- 4: emb[g] += x[i]*w[i]  (sorted batch, thread = column) --------
__global__ __launch_bounds__(256)
void wsum_kernel(const float* __restrict__ x, const long long* __restrict__ batch,
                 const float* __restrict__ e, const float* __restrict__ denom,
                 float* __restrict__ emb, int N) {
  __shared__ float wbuf[256];
  __shared__ int   gbuf[256];
  int c0 = blockIdx.x * 256;
  int t  = threadIdx.x;
  int cnt = N - c0; if (cnt > 256) cnt = 256;
  if (t < cnt) {
    int g = (int)batch[c0 + t];
    gbuf[t] = g;
    wbuf[t] = e[c0 + t] / denom[g];
  }
  __syncthreads();
  float accv = 0.0f;
  int curg = gbuf[0];
  for (int j = 0; j < cnt; ++j) {
    int g = gbuf[j];
    if (g != curg) {
      atomicAdd(&emb[(size_t)curg * H + t], accv);
      accv = 0.0f; curg = g;
    }
    accv += x[(size_t)(c0 + j) * H + t] * wbuf[j];
  }
  atomicAdd(&emb[(size_t)curg * H + t], accv);
}

// -------- 5: ctx = emb @ Wp + bp  (tiny, full f32) --------
__global__ __launch_bounds__(256)
void context_kernel(const float* __restrict__ emb, const float* __restrict__ Wp,
                    const float* __restrict__ bp, float* __restrict__ ctx) {
  __shared__ float row[H];
  int g = blockIdx.x, t = threadIdx.x;
  row[t] = emb[(size_t)g * H + t];
  __syncthreads();
  float sum = bp[t];
  #pragma unroll 8
  for (int k = 0; k < H; ++k) sum += row[k] * Wp[(size_t)k * H + t];
  ctx[(size_t)g * H + t] = sum;
}

// -------- 6: out[i] = ctx[batch[i]]  (float4) --------
__global__ __launch_bounds__(256)
void scatter_kernel(const float* __restrict__ ctx, const long long* __restrict__ batch,
                    float* __restrict__ out, int N) {
  int tid = blockIdx.x * 256 + threadIdx.x;
  int total = N * 64;                 // H/4 float4 per row
  if (tid < total) {
    int r = tid >> 6, c = tid & 63;
    int g = (int)batch[r];
    ((float4*)out)[(size_t)r * 64 + c] = ((const float4*)ctx)[(size_t)g * 64 + c];
  }
}

extern "C" void kernel_launch(void* const* d_in, const int* in_sizes, int n_in,
                              void* d_out, int out_size, void* d_ws, size_t ws_size,
                              hipStream_t stream) {
  const float*     x     = (const float*)d_in[0];
  const long long* batch = (const long long*)d_in[1];
  const float*     W1    = (const float*)d_in[2];
  const float*     b1    = (const float*)d_in[3];
  const float*     W2    = (const float*)d_in[4];
  const float*     b2    = (const float*)d_in[5];
  const float*     Wp    = (const float*)d_in[6];
  const float*     bp    = (const float*)d_in[7];
  float*           out   = (float*)d_out;

  int N  = in_sizes[1];          // batch has N elements
  int Np = (N + 3) & ~3;         // keep 16B alignment for downstream buffers

  float*    s     = (float*)d_ws;
  float*    e     = s + Np;
  unsigned* smax  = (unsigned*)(e + Np);
  float*    denom = (float*)(smax + G);
  float*    emb   = denom + G;
  float*    ctx   = emb + (size_t)G * H;

  // zero [smax | denom | emb] (contiguous)
  int zn = G + G + G * H;
  zero_kernel<<<(zn + 255) / 256, 256, 0, stream>>>((float*)smax, zn);

  int ntiles = (N + 15) / 16;
  gate_kernel<<<(ntiles + 7) / 8, 256, 0, stream>>>(x, W1, b1, W2, b2, s, N, ntiles);

  int nb = (N + 255) / 256;
  segmax_kernel<<<nb, 256, 0, stream>>>(s, batch, smax, N);
  expsum_kernel<<<nb, 256, 0, stream>>>(s, batch, smax, e, denom, N);
  wsum_kernel<<<nb, 256, 0, stream>>>(x, batch, e, denom, emb, N);
  context_kernel<<<G, 256, 0, stream>>>(emb, Wp, bp, ctx);

  long long total4 = (long long)N * 64;
  scatter_kernel<<<(int)((total4 + 255) / 256), 256, 0, stream>>>(ctx, batch, out, N);
}
